// RopeSelfAttention_79345225826639
// MI455X (gfx1250) — matrix operand order repaired and checked by reading, not compiled, not run on hardware
//
#include <hip/hip_runtime.h>

typedef __bf16 bf16_t;
typedef __attribute__((ext_vector_type(16))) __bf16 v16bf;
typedef __attribute__((ext_vector_type(8)))  __bf16 v8bf;
typedef __attribute__((ext_vector_type(8)))  float  v8f;

// ---------------------------------------------------------------------------
// CDNA5 async global->LDS copy (ASYNCcnt path) with safe fallback
// ---------------------------------------------------------------------------
__device__ __forceinline__ void wait_async0() {
#if __has_builtin(__builtin_amdgcn_s_wait_asynccnt)
  __builtin_amdgcn_s_wait_asynccnt(0);
#else
  asm volatile("s_wait_asynccnt 0" ::: "memory");
#endif
}

typedef int v4i_ld __attribute__((vector_size(16)));
typedef __attribute__((address_space(1))) v4i_ld* async_gptr;
typedef __attribute__((address_space(3))) v4i_ld* async_lptr;

template <typename T>
__device__ __forceinline__ void async_copy16(T* lds_dst, const T* gsrc) {
#if __has_builtin(__builtin_amdgcn_global_load_async_to_lds_b128)
  __builtin_amdgcn_global_load_async_to_lds_b128(
      (async_gptr)(void*)gsrc, (async_lptr)(void*)lds_dst, 0, 0);
#else
  *(uint4*)lds_dst = *(const uint4*)gsrc;
#endif
}

// ---------------------------------------------------------------------------
// WMMA wrapper: D = A(16x32 bf16) * B(32x16 bf16) + C(16x16 f32)
// ---------------------------------------------------------------------------
__device__ __forceinline__ v8f wmma_bf16(v16bf a, v16bf b, v8f c) {
  return __builtin_amdgcn_wmma_f32_16x16x32_bf16(
      /*neg_a=*/false, a, /*neg_b=*/false, b,
      /*c_mod=*/(short)0, c, /*reuse_a=*/false, /*reuse_b=*/false);
}

// ---------------------------------------------------------------------------
// Fragment builders (contiguous k = 16*(lane/16) + s order, wave32). For bf16 sources the
// per-lane elements are contiguous 16B runs -> compiler emits ds_load_b128.
// ---------------------------------------------------------------------------
template <typename T>
__device__ __forceinline__ v16bf frag_a_rm(const T* __restrict__ p, int ld) {
  const int lane = threadIdx.x & 31;
  const int m = lane & 15;
  const int kg = (lane >> 4) << 4;           // 0 or 16
  const T* r = p + (size_t)m * ld;
  v16bf a;
#pragma unroll
  for (int v = 0; v < 8; ++v) {
    const int k = kg + 2 * v;
    a[2 * v]     = (bf16_t)r[k];
    a[2 * v + 1] = (bf16_t)r[k + 1];
  }
  return a;
}

// B 32x16 (KxN), source stores N rows of K values: B(k,n) = p[n*ld + k]
template <typename T>
__device__ __forceinline__ v16bf frag_b_tr(const T* __restrict__ p, int ld) {
  const int lane = threadIdx.x & 31;
  const int n = lane & 15;
  const int kb = (lane >> 4) << 4;           // 0 or 16
  const T* r = p + (size_t)n * ld + kb;
  v16bf b;
#pragma unroll
  for (int v = 0; v < 8; ++v) {
    b[2 * v]     = (bf16_t)r[2 * v];
    b[2 * v + 1] = (bf16_t)r[2 * v + 1];
  }
  return b;
}

// B 32x16 (KxN) row-major source: B(k,n) = p[k*ld + n]
template <typename T>
__device__ __forceinline__ v16bf frag_b_rm(const T* __restrict__ p, int ld) {
  const int lane = threadIdx.x & 31;
  const int n = lane & 15;
  const int kb = (lane >> 4) << 4;
  v16bf b;
#pragma unroll
  for (int v = 0; v < 8; ++v) {
    b[2 * v]     = (bf16_t)p[(size_t)(kb + 2 * v) * ld + n];
    b[2 * v + 1] = (bf16_t)p[(size_t)(kb + 2 * v + 1) * ld + n];
  }
  return b;
}

__device__ __forceinline__ void store_c(float* __restrict__ p, int ld, v8f c) {
  const int lane = threadIdx.x & 31;
  const int n = lane & 15;
  const int mb = (lane >> 4) << 3;
#pragma unroll
  for (int r = 0; r < 8; ++r) p[(size_t)(mb + r) * ld + n] = c[r];
}

__device__ __forceinline__ void store_c_bias(float* __restrict__ p, int ld, v8f c,
                                             const float* __restrict__ bias) {
  const int lane = threadIdx.x & 31;
  const int n = lane & 15;
  const int mb = (lane >> 4) << 3;
  const float bv = bias[n];
#pragma unroll
  for (int r = 0; r < 8; ++r) p[(size_t)(mb + r) * ld + n] = c[r] + bv;
}

// ===========================================================================
// Kernel 0: one-shot f32 -> bf16 conversion (8 elems/thread)
// ===========================================================================
__global__ __launch_bounds__(256) void cvt_bf16_kernel(const float* __restrict__ src,
                                                       bf16_t* __restrict__ dst) {
  const size_t i = ((size_t)blockIdx.x * 256 + threadIdx.x) * 8;
  const float4 a = *(const float4*)(src + i);
  const float4 b = *(const float4*)(src + i + 4);
  v8bf o;
  o[0] = (bf16_t)a.x; o[1] = (bf16_t)a.y; o[2] = (bf16_t)a.z; o[3] = (bf16_t)a.w;
  o[4] = (bf16_t)b.x; o[5] = (bf16_t)b.y; o[6] = (bf16_t)b.z; o[7] = (bf16_t)b.w;
  *(v8bf*)(dst + i) = o;
}

// ===========================================================================
// Kernel 1: qkv = X[2048x2048 bf16] @ Wqkv^T[2048x6144 bf16], f32 out
// KT=64, double-buffered async bf16 LDS staging
// ===========================================================================
__global__ __launch_bounds__(256) void qkv_gemm_kernel(
    const bf16_t* __restrict__ X, const bf16_t* __restrict__ W, float* __restrict__ out) {
  constexpr int K = 2048, N = 6144, SLD = 72;   // 72 bf16 = 144B rows (16B aligned)
  __shared__ bf16_t sA[2][128 * SLD];
  __shared__ bf16_t sB[2][128 * SLD];
  const int m0 = blockIdx.y * 128;
  const int n0 = blockIdx.x * 128;
  const int wave = threadIdx.x >> 5;
  const int wr = (wave >> 1) * 32;
  const int wc = (wave & 1) * 64;

  auto load_tiles = [&](int buf, int k0) {
#pragma unroll
    for (int i = 0; i < 4; ++i) {
      const int fid = threadIdx.x + i * 256;   // 1024 b128 slots per 128x64 tile
      const int row = fid >> 3;
      const int c8 = (fid & 7) << 3;
      async_copy16(&sA[buf][row * SLD + c8], X + (size_t)(m0 + row) * K + k0 + c8);
      async_copy16(&sB[buf][row * SLD + c8], W + (size_t)(n0 + row) * K + k0 + c8);
    }
  };

  load_tiles(0, 0);
  wait_async0();
  __syncthreads();

  v8f acc[2][4] = {};
  int buf = 0;
  for (int k0 = 0; k0 < K; k0 += 64) {
    if (k0 + 64 < K) load_tiles(buf ^ 1, k0 + 64);  // overlap copy with WMMA
#pragma unroll
    for (int kk = 0; kk < 2; ++kk) {
      const v16bf a0 = frag_a_rm(&sA[buf][(wr + 0) * SLD + kk * 32], SLD);
      const v16bf a1 = frag_a_rm(&sA[buf][(wr + 16) * SLD + kk * 32], SLD);
#pragma unroll
      for (int j = 0; j < 4; ++j) {
        const v16bf b = frag_b_tr(&sB[buf][(wc + 16 * j) * SLD + kk * 32], SLD);
        acc[0][j] = wmma_bf16(a0, b, acc[0][j]);
        acc[1][j] = wmma_bf16(a1, b, acc[1][j]);
      }
    }
    wait_async0();
    __syncthreads();
    buf ^= 1;
  }
#pragma unroll
  for (int i = 0; i < 2; ++i)
#pragma unroll
    for (int j = 0; j < 4; ++j)
      store_c(out + (size_t)(m0 + wr + 16 * i) * N + (n0 + wc + 16 * j), N, acc[i][j]);
}

// ===========================================================================
// Kernel 2: QKNorm (fp32 RMS) + RoPE, split qkv -> q/k/v bf16 [H][L][D]
// ===========================================================================
__global__ __launch_bounds__(256) void qknorm_rope_kernel(
    const float* __restrict__ qkv, const float* __restrict__ pe,
    const float* __restrict__ q_scale, const float* __restrict__ k_scale,
    bf16_t* __restrict__ qb, bf16_t* __restrict__ kb, bf16_t* __restrict__ vb) {
  const int wave = threadIdx.x >> 5;
  const int lane = threadIdx.x & 31;
  const int row = blockIdx.x * 8 + wave;   // 0 .. H*L-1
  const int h = row >> 11;
  const int l = row & 2047;

  const size_t base = (size_t)l * 6144 + (size_t)h * 128 + lane * 4;
  const float4 qv = *(const float4*)(qkv + base);
  const float4 kv = *(const float4*)(qkv + base + 2048);
  const float4 vv = *(const float4*)(qkv + base + 4096);

  float sq = qv.x * qv.x + qv.y * qv.y + qv.z * qv.z + qv.w * qv.w;
  float sk = kv.x * kv.x + kv.y * kv.y + kv.z * kv.z + kv.w * kv.w;
#pragma unroll
  for (int m = 1; m < 32; m <<= 1) {
    sq += __shfl_xor(sq, m, 32);
    sk += __shfl_xor(sk, m, 32);
  }
  const float rq = rsqrtf(sq * (1.0f / 128.0f) + 1e-6f);
  const float rk = rsqrtf(sk * (1.0f / 128.0f) + 1e-6f);

  const float4 qs = *(const float4*)(q_scale + lane * 4);
  const float4 ks = *(const float4*)(k_scale + lane * 4);
  const float q0 = qv.x * rq * qs.x, q1 = qv.y * rq * qs.y;
  const float q2 = qv.z * rq * qs.z, q3 = qv.w * rq * qs.w;
  const float k0 = kv.x * rk * ks.x, k1 = kv.y * rk * ks.y;
  const float k2 = kv.z * rk * ks.z, k3 = kv.w * rk * ks.w;

  const float* pb = pe + (size_t)l * 256 + (size_t)lane * 8;
  const float oq0 = pb[0] * q0 + pb[1] * q1, oq1 = pb[2] * q0 + pb[3] * q1;
  const float oq2 = pb[4] * q2 + pb[5] * q3, oq3 = pb[6] * q2 + pb[7] * q3;
  const float ok0 = pb[0] * k0 + pb[1] * k1, ok1 = pb[2] * k0 + pb[3] * k1;
  const float ok2 = pb[4] * k2 + pb[5] * k3, ok3 = pb[6] * k2 + pb[7] * k3;

  const size_t ob = ((size_t)h * 2048 + l) * 128 + lane * 4;
  qb[ob + 0] = (bf16_t)oq0; qb[ob + 1] = (bf16_t)oq1;
  qb[ob + 2] = (bf16_t)oq2; qb[ob + 3] = (bf16_t)oq3;
  kb[ob + 0] = (bf16_t)ok0; kb[ob + 1] = (bf16_t)ok1;
  kb[ob + 2] = (bf16_t)ok2; kb[ob + 3] = (bf16_t)ok3;
  vb[ob + 0] = (bf16_t)vv.x; vb[ob + 1] = (bf16_t)vv.y;
  vb[ob + 2] = (bf16_t)vv.z; vb[ob + 3] = (bf16_t)vv.w;
}

// ===========================================================================
// Kernel 3: flash attention per head, double-buffered async K/V staging,
// bf16 P tile (no f32->bf16 cvt on the reload path)
// ===========================================================================
__global__ __launch_bounds__(128) void attention_kernel(
    const bf16_t* __restrict__ Q, const bf16_t* __restrict__ Kmat,
    const bf16_t* __restrict__ Vmat, bf16_t* __restrict__ O) {
  constexpr int SKV = 136;  // bf16 stride for 32x128 K/V tiles
  constexpr int SPB = 40;   // bf16 stride for 16x32 P tile
  __shared__ bf16_t sK[2][32 * SKV];
  __shared__ bf16_t sV[2][32 * SKV];
  __shared__ bf16_t sP[4][16 * SPB];

  const int h = blockIdx.y;
  const int wave = threadIdx.x >> 5;
  const int lane = threadIdx.x & 31;
  const int qrow0 = blockIdx.x * 64 + wave * 16;
  const int nn = lane & 15;
  const int mb = (lane >> 4) << 3;

  const bf16_t* Qh = Q + (size_t)h * 2048 * 128;
  const bf16_t* Kh = Kmat + (size_t)h * 2048 * 128;
  const bf16_t* Vh = Vmat + (size_t)h * 2048 * 128;

  auto load_kv = [&](int buf, int kt) {
#pragma unroll
    for (int i = 0; i < 4; ++i) {
      const int fid = threadIdx.x + i * 128;  // 512 b128 slots per tile
      const int row = fid >> 4;
      const int c8 = (fid & 15) << 3;
      async_copy16(&sK[buf][row * SKV + c8], Kh + (size_t)(kt + row) * 128 + c8);
      async_copy16(&sV[buf][row * SKV + c8], Vh + (size_t)(kt + row) * 128 + c8);
    }
  };

  // resident Q fragments: 16 rows x 128 d as 4 A-fragments
  v16bf qa[4];
#pragma unroll
  for (int c = 0; c < 4; ++c)
    qa[c] = frag_a_rm(Qh + (size_t)qrow0 * 128 + c * 32, 128);

  v8f o[8] = {};
  float mstat[8], lstat[8];
#pragma unroll
  for (int r = 0; r < 8; ++r) { mstat[r] = -1.0e30f; lstat[r] = 0.0f; }

  const float scale = 0.08838834764831845f;  // 1/sqrt(128)

  load_kv(0, 0);
  wait_async0();
  __syncthreads();

  int buf = 0;
  for (int kt = 0; kt < 2048; kt += 32) {
    if (kt + 32 < 2048) load_kv(buf ^ 1, kt + 32);

    // S = Q * K^T : two 16x16 f32 fragments (keys kt..kt+15, kt+16..kt+31)
    v8f s0 = {}, s1 = {};
#pragma unroll
    for (int c = 0; c < 4; ++c) {
      s0 = wmma_bf16(qa[c], frag_b_tr(&sK[buf][0 * SKV + c * 32], SKV), s0);
      s1 = wmma_bf16(qa[c], frag_b_tr(&sK[buf][16 * SKV + c * 32], SKV), s1);
    }

    // online softmax (a row lives in one 16-lane half: 4 xor-shuffle steps)
#pragma unroll
    for (int r = 0; r < 8; ++r) {
      const float v0 = s0[r] * scale, v1 = s1[r] * scale;
      float mx = fmaxf(v0, v1);
#pragma unroll
      for (int msk = 1; msk < 16; msk <<= 1) mx = fmaxf(mx, __shfl_xor(mx, msk, 32));
      const float nm = fmaxf(mstat[r], mx);
      const float corr = __expf(mstat[r] - nm);
      const float e0 = __expf(v0 - nm), e1 = __expf(v1 - nm);
      float rs = e0 + e1;
#pragma unroll
      for (int msk = 1; msk < 16; msk <<= 1) rs += __shfl_xor(rs, msk, 32);
      lstat[r] = lstat[r] * corr + rs;
      mstat[r] = nm;
#pragma unroll
      for (int f = 0; f < 8; ++f) o[f][r] *= corr;
      sP[wave][(mb + r) * SPB + nn] = (bf16_t)e0;   // C-frag -> A-frag re-layout
      sP[wave][(mb + r) * SPB + 16 + nn] = (bf16_t)e1;
    }

    // O += P(16x32) * V(32x128)
    const v16bf pa = frag_a_rm(&sP[wave][0], SPB);
#pragma unroll
    for (int f = 0; f < 8; ++f)
      o[f] = wmma_bf16(pa, frag_b_rm(&sV[buf][f * 16], SKV), o[f]);

    wait_async0();
    __syncthreads();
    buf ^= 1;
  }

  // normalize and emit O in token-major [L][DIM] bf16 layout
#pragma unroll
  for (int r = 0; r < 8; ++r) {
    const float inv = 1.0f / lstat[r];
#pragma unroll
    for (int f = 0; f < 8; ++f) o[f][r] *= inv;
  }
  bf16_t* Ob = O + (size_t)qrow0 * 2048 + h * 128;
#pragma unroll
  for (int f = 0; f < 8; ++f)
#pragma unroll
    for (int r = 0; r < 8; ++r)
      Ob[(size_t)(mb + r) * 2048 + f * 16 + nn] = (bf16_t)o[f][r];
}

// ===========================================================================
// Kernel 4: out = O[2048x2048 bf16] @ Wproj^T[bf16] + b  (f32 out), KT=64
// ===========================================================================
__global__ __launch_bounds__(256) void proj_gemm_kernel(
    const bf16_t* __restrict__ A, const bf16_t* __restrict__ W,
    const float* __restrict__ bias, float* __restrict__ out) {
  constexpr int K = 2048, N = 2048, SLD = 72;
  __shared__ bf16_t sA[2][128 * SLD];
  __shared__ bf16_t sB[2][128 * SLD];
  const int m0 = blockIdx.y * 128;
  const int n0 = blockIdx.x * 128;
  const int wave = threadIdx.x >> 5;
  const int wr = (wave >> 1) * 32;
  const int wc = (wave & 1) * 64;

  auto load_tiles = [&](int buf, int k0) {
#pragma unroll
    for (int i = 0; i < 4; ++i) {
      const int fid = threadIdx.x + i * 256;
      const int row = fid >> 3;
      const int c8 = (fid & 7) << 3;
      async_copy16(&sA[buf][row * SLD + c8], A + (size_t)(m0 + row) * K + k0 + c8);
      async_copy16(&sB[buf][row * SLD + c8], W + (size_t)(n0 + row) * K + k0 + c8);
    }
  };

  load_tiles(0, 0);
  wait_async0();
  __syncthreads();

  v8f acc[2][4] = {};
  int buf = 0;
  for (int k0 = 0; k0 < K; k0 += 64) {
    if (k0 + 64 < K) load_tiles(buf ^ 1, k0 + 64);
#pragma unroll
    for (int kk = 0; kk < 2; ++kk) {
      const v16bf a0 = frag_a_rm(&sA[buf][(wr + 0) * SLD + kk * 32], SLD);
      const v16bf a1 = frag_a_rm(&sA[buf][(wr + 16) * SLD + kk * 32], SLD);
#pragma unroll
      for (int j = 0; j < 4; ++j) {
        const v16bf b = frag_b_tr(&sB[buf][(wc + 16 * j) * SLD + kk * 32], SLD);
        acc[0][j] = wmma_bf16(a0, b, acc[0][j]);
        acc[1][j] = wmma_bf16(a1, b, acc[1][j]);
      }
    }
    wait_async0();
    __syncthreads();
    buf ^= 1;
  }
#pragma unroll
  for (int i = 0; i < 2; ++i)
#pragma unroll
    for (int j = 0; j < 4; ++j)
      store_c_bias(out + (size_t)(m0 + wr + 16 * i) * N + (n0 + wc + 16 * j), N,
                   acc[i][j], bias + n0 + wc + 16 * j);
}

// ===========================================================================
extern "C" void kernel_launch(void* const* d_in, const int* in_sizes, int n_in,
                              void* d_out, int out_size, void* d_ws, size_t ws_size,
                              hipStream_t stream) {
  (void)in_sizes; (void)n_in; (void)out_size; (void)ws_size;
  const float* x       = (const float*)d_in[0];
  const float* pe      = (const float*)d_in[1];
  const float* w_qkv   = (const float*)d_in[2];
  const float* q_scale = (const float*)d_in[3];
  const float* k_scale = (const float*)d_in[4];
  const float* w_proj  = (const float*)d_in[5];
  const float* b_proj  = (const float*)d_in[6];
  float* out = (float*)d_out;

  // workspace layout (~120 MB)
  char* ws = (char*)d_ws;
  float* qkv_f32 = (float*)ws;                               // 48 MB
  bf16_t* qb = (bf16_t*)(ws + (size_t)2048 * 6144 * 4);      // 8 MB each
  bf16_t* kb = qb + (size_t)16 * 2048 * 128;
  bf16_t* vb = kb + (size_t)16 * 2048 * 128;
  bf16_t* ob = vb + (size_t)16 * 2048 * 128;                 // 8 MB
  bf16_t* xb = ob + (size_t)2048 * 2048;                     // 8 MB
  bf16_t* wqkvb = xb + (size_t)2048 * 2048;                  // 24 MB
  bf16_t* wprojb = wqkvb + (size_t)6144 * 2048;              // 8 MB

  // one-shot bf16 conversion of operands (memory-cheap, removes cvt from hot loops)
  cvt_bf16_kernel<<<dim3(2048), dim3(256), 0, stream>>>(x, xb);
  cvt_bf16_kernel<<<dim3(6144), dim3(256), 0, stream>>>(w_qkv, wqkvb);
  cvt_bf16_kernel<<<dim3(2048), dim3(256), 0, stream>>>(w_proj, wprojb);

  qkv_gemm_kernel<<<dim3(48, 16, 1), dim3(256, 1, 1), 0, stream>>>(xb, wqkvb, qkv_f32);
  qknorm_rope_kernel<<<dim3(4096, 1, 1), dim3(256, 1, 1), 0, stream>>>(
      qkv_f32, pe, q_scale, k_scale, qb, kb, vb);
  attention_kernel<<<dim3(32, 16, 1), dim3(128, 1, 1), 0, stream>>>(qb, kb, vb, ob);
  proj_gemm_kernel<<<dim3(16, 16, 1), dim3(256, 1, 1), 0, stream>>>(ob, wprojb, b_proj, out);
}
